// TextMatchBert_75737453298239
// MI455X (gfx1250) — compile-verified, matrix operand order checked by reading, not run on hardware
//
#include <hip/hip_runtime.h>
#include <hip/hip_bf16.h>
#include <math.h>

// ---------------- problem constants (match reference) ----------------
#define Bb   32
#define Ss   128
#define Tn   (Bb*Ss)      // 4096 token rows
#define Hd   768
#define FFd  3072
#define NHd  12
#define DHd  64
#define Ll   12
#define NLd  10000

typedef __attribute__((ext_vector_type(16))) _Float16 v16h;
typedef __attribute__((ext_vector_type(8)))  _Float16 v8h;
typedef __attribute__((ext_vector_type(8)))  float    v8f;

#define LDS_PAD 40   // halfs per LDS row; 80B = 16B-aligned rows for b128 frag loads

__device__ __forceinline__ float gelu_tanh(float x) {
    // jax.nn.gelu default (tanh approximation)
    float x3 = x * x * x;
    return 0.5f * x * (1.0f + tanhf(0.7978845608028654f * (x + 0.044715f * x3)));
}

// =====================================================================
// Generic WMMA GEMM:  C[M,N] = A[M,K] * op(B) (+bias)(+gelu)
//   A: fp32 row-major, lda
//   B: fp32; TRANSB=0 -> [K,N] row-major; TRANSB=1 -> [N,K] row-major
//   fp32->fp16 conversion while staging into double-buffered LDS.
//   Batched via blockIdx.z with outer/inner stride decomposition.
// Block: 256 threads (8 waves). Tile: 128(M) x 128(N) x 32(K).
// Each wave: 32x64 -> 2 A frags x 4 B frags -> 8 v_wmma per K step.
// Pipeline: global loads for tile k+1 issued before WMMAs on tile k;
// one barrier per K step.
// =====================================================================
template<bool TRANSB, int EPI>   // EPI: 0=none 1=+bias 2=+bias,gelu
__global__ __launch_bounds__(256)
void gemm_wmma_kernel(const float* __restrict__ A, const float* __restrict__ Bm,
                      const float* __restrict__ bias, float* __restrict__ C,
                      int M, int N, int K, int lda, int ldb, int ldc,
                      long sAo, long sAi, long sBo, long sBi, long sCo, long sCi,
                      int inner)
{
    __shared__ __align__(16) _Float16 As[2][128][LDS_PAD];
    __shared__ __align__(16) _Float16 Bs[2][128][LDS_PAD];

    const int z  = blockIdx.z;
    const int zo = z / inner;
    const int zi = z - zo * inner;
    A  += (long)zo * sAo + (long)zi * sAi;
    Bm += (long)zo * sBo + (long)zi * sBi;
    C  += (long)zo * sCo + (long)zi * sCi;

    const int tid  = threadIdx.x;
    const int lane = tid & 31;
    const int wid  = tid >> 5;
    const int wm   = wid & 3;       // wave row   (4 x 32 rows)
    const int wn   = wid >> 2;      // wave col   (2 x 64 cols)
    const int hs   = lane >> 4;     // lane-half select
    const int l16  = lane & 15;

    const int m0 = blockIdx.y * 128;
    const int n0 = blockIdx.x * 128;

    // block-uniform full-tile predicates -> branch-free hot path
    const bool fullM = (m0 + 128 <= M);
    const bool fullN = (n0 + 128 <= N);

    float4 ra[4], rb[4];   // in-flight global tiles (registers)

    auto load_global = [&](int k0) {
        // A tile: 128x32, float4 per thread x4
        #pragma unroll
        for (int i = 0; i < 4; ++i) {
            int f4 = tid + i * 256;
            int r  = f4 >> 3;
            int c4 = (f4 & 7) << 2;
            int gm = m0 + r;
            if (fullM) {
                ra[i] = *(const float4*)(A + (long)gm * lda + k0 + c4);
            } else {
                ra[i] = make_float4(0.f, 0.f, 0.f, 0.f);
                if (gm < M) ra[i] = *(const float4*)(A + (long)gm * lda + k0 + c4);
            }
        }
        if (TRANSB) {
            #pragma unroll
            for (int i = 0; i < 4; ++i) {
                int f4 = tid + i * 256;
                int r  = f4 >> 3;
                int c4 = (f4 & 7) << 2;
                int gn = n0 + r;
                if (fullN) {
                    rb[i] = *(const float4*)(Bm + (long)gn * ldb + k0 + c4);
                } else {
                    rb[i] = make_float4(0.f, 0.f, 0.f, 0.f);
                    if (gn < N) rb[i] = *(const float4*)(Bm + (long)gn * ldb + k0 + c4);
                }
            }
        } else {
            #pragma unroll
            for (int i = 0; i < 4; ++i) {
                int f4 = tid + i * 256;
                int kr = f4 >> 5;
                int n4 = (f4 & 31) << 2;
                int gn = n0 + n4;
                if (fullN) {
                    rb[i] = *(const float4*)(Bm + (long)(k0 + kr) * ldb + gn);
                } else {
                    rb[i] = make_float4(0.f, 0.f, 0.f, 0.f);
                    if (gn < N) rb[i] = *(const float4*)(Bm + (long)(k0 + kr) * ldb + gn);
                }
            }
        }
    };

    auto store_lds = [&](int buf) {
        #pragma unroll
        for (int i = 0; i < 4; ++i) {
            int f4 = tid + i * 256;
            int r  = f4 >> 3;
            int c4 = (f4 & 7) << 2;
            As[buf][r][c4 + 0] = (_Float16)ra[i].x;
            As[buf][r][c4 + 1] = (_Float16)ra[i].y;
            As[buf][r][c4 + 2] = (_Float16)ra[i].z;
            As[buf][r][c4 + 3] = (_Float16)ra[i].w;
        }
        if (TRANSB) {
            #pragma unroll
            for (int i = 0; i < 4; ++i) {
                int f4 = tid + i * 256;
                int r  = f4 >> 3;
                int c4 = (f4 & 7) << 2;
                Bs[buf][r][c4 + 0] = (_Float16)rb[i].x;
                Bs[buf][r][c4 + 1] = (_Float16)rb[i].y;
                Bs[buf][r][c4 + 2] = (_Float16)rb[i].z;
                Bs[buf][r][c4 + 3] = (_Float16)rb[i].w;
            }
        } else {
            #pragma unroll
            for (int i = 0; i < 4; ++i) {
                int f4 = tid + i * 256;
                int kr = f4 >> 5;
                int n4 = (f4 & 31) << 2;
                Bs[buf][n4 + 0][kr] = (_Float16)rb[i].x;
                Bs[buf][n4 + 1][kr] = (_Float16)rb[i].y;
                Bs[buf][n4 + 2][kr] = (_Float16)rb[i].z;
                Bs[buf][n4 + 3][kr] = (_Float16)rb[i].w;
            }
        }
    };

    v8f acc[2][4] = {};

    // prologue: stage tile 0
    load_global(0);
    store_lds(0);
    __syncthreads();

    const int nk = K >> 5;
    for (int kt = 0; kt < nk; ++kt) {
        const int cur = kt & 1;

        // issue next tile's global loads early (hide under WMMA)
        if (kt + 1 < nk) load_global((kt + 1) << 5);

        // prefetch two tiles ahead (global_prefetch_b8)
        if (kt + 2 < nk) {
            int kp0 = (kt + 2) << 5;
            int pr  = tid >> 1;
            if (m0 + pr < M)
                __builtin_prefetch(A + (long)(m0 + pr) * lda + kp0 + (tid & 1) * 16, 0, 1);
            if (TRANSB) {
                if (n0 + pr < N)
                    __builtin_prefetch(Bm + (long)(n0 + pr) * ldb + kp0 + (tid & 1) * 16, 0, 1);
            } else {
                int kr = tid >> 3;
                int nn = (tid & 7) * 16;
                if (n0 + nn < N)
                    __builtin_prefetch(Bm + (long)(kp0 + kr) * ldb + n0 + nn, 0, 1);
            }
        }

        // ---- fragment loads: 2x ds_load_b128 per fragment ----
        // f16 A layout per lane: halfs [0..7] = K(hs*8 + 0..7),
        //                        halfs [8..15] = K(hs*8 + 16..23)
        v16h af[2], bf[4];
        #pragma unroll
        for (int fm = 0; fm < 2; ++fm) {
            const _Float16* ap = &As[cur][wm * 32 + fm * 16 + l16][hs * 8];
            v8h lo = *(const v8h*)(ap);
            v8h hi = *(const v8h*)(ap + 16);
            af[fm] = __builtin_shufflevector(lo, hi, 0,1,2,3,4,5,6,7,8,9,10,11,12,13,14,15);
        }
        #pragma unroll
        for (int fn = 0; fn < 4; ++fn) {
            const _Float16* bp = &Bs[cur][wn * 64 + fn * 16 + l16][hs * 8];
            v8h lo = *(const v8h*)(bp);
            v8h hi = *(const v8h*)(bp + 16);
            bf[fn] = __builtin_shufflevector(lo, hi, 0,1,2,3,4,5,6,7,8,9,10,11,12,13,14,15);
        }

        // ---- 8 WMMAs per K step ----
        #pragma unroll
        for (int fm = 0; fm < 2; ++fm)
            #pragma unroll
            for (int fn = 0; fn < 4; ++fn)
                acc[fm][fn] = __builtin_amdgcn_wmma_f32_16x16x32_f16(
                    false, af[fm], false, bf[fn], (short)0, acc[fm][fn], false, false);

        // convert + store next tile into the other buffer; single barrier
        if (kt + 1 < nk) {
            store_lds(cur ^ 1);
            __syncthreads();
        }
    }

    // ---- epilogue: bias / gelu, C layout per 16x16 f32 D mapping ----
    #pragma unroll
    for (int fm = 0; fm < 2; ++fm) {
        #pragma unroll
        for (int fn = 0; fn < 4; ++fn) {
            int col = n0 + wn * 64 + fn * 16 + l16;
            if (col >= N) continue;
            float bv = (EPI >= 1 && bias) ? bias[col] : 0.f;
            #pragma unroll
            for (int r = 0; r < 8; ++r) {
                int row = m0 + wm * 32 + fm * 16 + hs * 8 + r;
                if (row >= M) continue;
                float v = acc[fm][fn][r] + bv;
                if (EPI == 2) v = gelu_tanh(v);
                C[(long)row * ldc + col] = v;
            }
        }
    }
}

// =====================================================================
// Fused embedding gather + LayerNorm (one block per token row, H=768)
// =====================================================================
__global__ __launch_bounds__(256)
void embed_ln_kernel(const int* __restrict__ ids, const int* __restrict__ types,
                     const float* __restrict__ we, const float* __restrict__ pe,
                     const float* __restrict__ te, const float* __restrict__ g,
                     const float* __restrict__ bb, float* __restrict__ out)
{
    __shared__ float red[256];
    int row = blockIdx.x;
    int s   = row % Ss;
    int tid = threadIdx.x;
    int id  = ids[row];
    int ty  = types[row];

    float v[3];
    float sum = 0.f;
    #pragma unroll
    for (int j = 0; j < 3; ++j) {
        int c = tid + j * 256;
        float t = we[(long)id * Hd + c] + pe[(long)s * Hd + c] + te[(long)ty * Hd + c];
        v[j] = t; sum += t;
    }
    red[tid] = sum; __syncthreads();
    for (int st = 128; st > 0; st >>= 1) { if (tid < st) red[tid] += red[tid + st]; __syncthreads(); }
    float mu = red[0] / (float)Hd; __syncthreads();

    float s2 = 0.f;
    #pragma unroll
    for (int j = 0; j < 3; ++j) { float d = v[j] - mu; s2 += d * d; }
    red[tid] = s2; __syncthreads();
    for (int st = 128; st > 0; st >>= 1) { if (tid < st) red[tid] += red[tid + st]; __syncthreads(); }
    float inv = rsqrtf(red[0] / (float)Hd + 1e-12f);

    #pragma unroll
    for (int j = 0; j < 3; ++j) {
        int c = tid + j * 256;
        out[(long)row * Hd + c] = (v[j] - mu) * inv * g[c] + bb[c];
    }
}

// =====================================================================
// LayerNorm(x + delta) -> out (H=768 rows), post-LN residual
// =====================================================================
__global__ __launch_bounds__(256)
void ln_residual_kernel(const float* __restrict__ x, const float* __restrict__ delta,
                        const float* __restrict__ g, const float* __restrict__ bb,
                        float* __restrict__ out)
{
    __shared__ float red[256];
    long row = blockIdx.x;
    int  tid = threadIdx.x;
    const float* xr = x + row * Hd;
    const float* dr = delta + row * Hd;

    float v[3];
    float sum = 0.f;
    #pragma unroll
    for (int j = 0; j < 3; ++j) {
        int c = tid + j * 256;
        float t = xr[c] + dr[c];
        v[j] = t; sum += t;
    }
    red[tid] = sum; __syncthreads();
    for (int st = 128; st > 0; st >>= 1) { if (tid < st) red[tid] += red[tid + st]; __syncthreads(); }
    float mu = red[0] / (float)Hd; __syncthreads();

    float s2 = 0.f;
    #pragma unroll
    for (int j = 0; j < 3; ++j) { float d = v[j] - mu; s2 += d * d; }
    red[tid] = s2; __syncthreads();
    for (int st = 128; st > 0; st >>= 1) { if (tid < st) red[tid] += red[tid + st]; __syncthreads(); }
    float inv = rsqrtf(red[0] / (float)Hd + 1e-12f);

    float* orow = out + row * Hd;
    #pragma unroll
    for (int j = 0; j < 3; ++j) {
        int c = tid + j * 256;
        orow[c] = (v[j] - mu) * inv * g[c] + bb[c];
    }
}

// =====================================================================
// Masked softmax over S=128 keys; one block (128 threads) per score row.
// =====================================================================
__global__ __launch_bounds__(128)
void softmax_mask_kernel(float* __restrict__ scores, const int* __restrict__ ids, float scale)
{
    __shared__ float red[128];
    long row = blockIdx.x;                 // B*NH*S rows
    int  b   = (int)(row / (NHd * Ss));
    float* p = scores + row * Ss;
    int tid  = threadIdx.x;

    int   id = ids[b * Ss + tid];
    float v  = p[tid] * scale + (id > 0 ? 0.f : -10000.f);

    red[tid] = v; __syncthreads();
    for (int st = 64; st > 0; st >>= 1) { if (tid < st) red[tid] = fmaxf(red[tid], red[tid + st]); __syncthreads(); }
    float mx = red[0]; __syncthreads();

    float e = __expf(v - mx);
    red[tid] = e; __syncthreads();
    for (int st = 64; st > 0; st >>= 1) { if (tid < st) red[tid] += red[tid + st]; __syncthreads(); }
    p[tid] = e / red[0];
}

// =====================================================================
// Small helpers: CLS extract, row L2 norms, cosine scale, argmax
// =====================================================================
__global__ void cls_extract_kernel(const float* __restrict__ x, float* __restrict__ cls)
{
    int i = blockIdx.x * 256 + threadIdx.x;     // < B*H
    int b = i / Hd, c = i - b * Hd;
    cls[i] = x[(long)b * Ss * Hd + c];
}

__global__ __launch_bounds__(256)
void row_norm_kernel(const float* __restrict__ X, float* __restrict__ out, int cols)
{
    __shared__ float red[256];
    long row = blockIdx.x;
    int  tid = threadIdx.x;
    const float* p = X + row * cols;
    float s = 0.f;
    for (int c = tid; c < cols; c += 256) { float v = p[c]; s += v * v; }
    red[tid] = s; __syncthreads();
    for (int st = 128; st > 0; st >>= 1) { if (tid < st) red[tid] += red[tid + st]; __syncthreads(); }
    if (tid == 0) out[row] = sqrtf(red[0]);
}

__global__ void cos_scale_kernel(float* __restrict__ num, const float* __restrict__ cn,
                                 const float* __restrict__ ln)
{
    long i = (long)blockIdx.x * 256 + threadIdx.x;
    if (i >= (long)Bb * NLd) return;
    int b = (int)(i / NLd), n = (int)(i - (long)b * NLd);
    num[i] = num[i] / (fmaxf(cn[b], 1e-8f) * fmaxf(ln[n], 1e-8f));
}

__global__ __launch_bounds__(256)
void argmax_kernel(const float* __restrict__ cosm, float* __restrict__ out_ids)
{
    __shared__ float rv[256];
    __shared__ int   ri[256];
    int b   = blockIdx.x;
    int tid = threadIdx.x;
    const float* row = cosm + (long)b * NLd;
    float best = -1e30f; int bi = 0;
    for (int i = tid; i < NLd; i += 256) {
        float v = row[i];
        if (v > best) { best = v; bi = i; }
    }
    rv[tid] = best; ri[tid] = bi; __syncthreads();
    for (int st = 128; st > 0; st >>= 1) {
        if (tid < st) {
            if (rv[tid + st] > rv[tid] ||
                (rv[tid + st] == rv[tid] && ri[tid + st] < ri[tid])) {
                rv[tid] = rv[tid + st]; ri[tid] = ri[tid + st];
            }
        }
        __syncthreads();
    }
    if (tid == 0) out_ids[b] = (float)ri[0];
}

// =====================================================================
// Host orchestration
// =====================================================================
static inline void launch_gemm_nn(const float* A, const float* B, const float* bias, float* C,
                                  int M, int N, int K, int lda, int ldb, int ldc,
                                  int epi, hipStream_t st)
{
    dim3 g((N + 127) / 128, (M + 127) / 128, 1), blk(256);
    if (epi == 0)
        gemm_wmma_kernel<false, 0><<<g, blk, 0, st>>>(A, B, bias, C, M, N, K, lda, ldb, ldc, 0, 0, 0, 0, 0, 0, 1);
    else if (epi == 1)
        gemm_wmma_kernel<false, 1><<<g, blk, 0, st>>>(A, B, bias, C, M, N, K, lda, ldb, ldc, 0, 0, 0, 0, 0, 0, 1);
    else
        gemm_wmma_kernel<false, 2><<<g, blk, 0, st>>>(A, B, bias, C, M, N, K, lda, ldb, ldc, 0, 0, 0, 0, 0, 0, 1);
}

extern "C" void kernel_launch(void* const* d_in, const int* in_sizes, int n_in,
                              void* d_out, int out_size, void* d_ws, size_t ws_size,
                              hipStream_t stream)
{
    const int*   input_ids  = (const int*)  d_in[0];
    const int*   input_type = (const int*)  d_in[1];
    const float* word_emb   = (const float*)d_in[2];
    const float* pos_emb    = (const float*)d_in[3];
    const float* type_emb   = (const float*)d_in[4];
    const float* emb_ln_g   = (const float*)d_in[5];
    const float* emb_ln_b   = (const float*)d_in[6];
    const float* Wq = (const float*)d_in[7],  *bq = (const float*)d_in[8];
    const float* Wk = (const float*)d_in[9],  *bk = (const float*)d_in[10];
    const float* Wv = (const float*)d_in[11], *bv = (const float*)d_in[12];
    const float* Wo = (const float*)d_in[13], *bo = (const float*)d_in[14];
    const float* ln1_g = (const float*)d_in[15], *ln1_b = (const float*)d_in[16];
    const float* W1 = (const float*)d_in[17], *b1 = (const float*)d_in[18];
    const float* W2 = (const float*)d_in[19], *b2 = (const float*)d_in[20];
    const float* ln2_g = (const float*)d_in[21], *ln2_b = (const float*)d_in[22];
    const float* label_map = (const float*)d_in[23];

    float* out = (float*)d_out;     // [B*NL] cosine, then [B] top-1 ids (as float)
    float* ws  = (float*)d_ws;

    // workspace layout (floats), with reuse: q<->ctx, k<->gemm-out, big = scores/ffn-hidden
    float* xw   = ws;                      // T*H
    float* qw   = xw  + (long)Tn * Hd;     // T*H   (q, then ctx)
    float* kw   = qw  + (long)Tn * Hd;     // T*H   (k, then gemm-out for residuals)
    float* vw   = kw  + (long)Tn * Hd;     // T*H
    float* big  = vw  + (long)Tn * Hd;     // T*FF  (scores [B*NH*S*S] then ffn hidden)
    float* clsw = big + (long)Tn * FFd;    // B*H
    float* cnw  = clsw + (long)Bb * Hd;    // B
    float* lnw  = cnw  + Bb;               // NL

    dim3 blk256(256);

    // ---- embeddings + LN ----
    embed_ln_kernel<<<dim3(Tn), blk256, 0, stream>>>(
        input_ids, input_type, word_emb, pos_emb, type_emb, emb_ln_g, emb_ln_b, xw);

    const float scale = 0.125f;   // 1/sqrt(64)

    for (int l = 0; l < Ll; ++l) {
        const float* wq = Wq + (long)l * Hd * Hd;  const float* bql = bq + (long)l * Hd;
        const float* wk = Wk + (long)l * Hd * Hd;  const float* bkl = bk + (long)l * Hd;
        const float* wv = Wv + (long)l * Hd * Hd;  const float* bvl = bv + (long)l * Hd;
        const float* wo = Wo + (long)l * Hd * Hd;  const float* bol = bo + (long)l * Hd;
        const float* w1 = W1 + (long)l * Hd * FFd; const float* b1l = b1 + (long)l * FFd;
        const float* w2 = W2 + (long)l * FFd * Hd; const float* b2l = b2 + (long)l * Hd;

        // Q, K, V projections (T x H) = x @ W + b
        launch_gemm_nn(xw, wq, bql, qw, Tn, Hd, Hd, Hd, Hd, Hd, 1, stream);
        launch_gemm_nn(xw, wk, bkl, kw, Tn, Hd, Hd, Hd, Hd, Hd, 1, stream);
        launch_gemm_nn(xw, wv, bvl, vw, Tn, Hd, Hd, Hd, Hd, Hd, 1, stream);

        // scores[b,nh,q,k] = Q . K^T   (batched over B*NH, TRANSB)
        {
            dim3 g(1, 1, Bb * NHd);
            gemm_wmma_kernel<true, 0><<<g, blk256, 0, stream>>>(
                qw, kw, nullptr, big,
                Ss, Ss, DHd, Hd, Hd, Ss,
                (long)Ss * Hd, (long)DHd,                 // A: outer=b, inner=nh
                (long)Ss * Hd, (long)DHd,                 // B
                (long)NHd * Ss * Ss, (long)Ss * Ss,       // C
                NHd);
        }
        // masked softmax
        softmax_mask_kernel<<<dim3(Bb * NHd * Ss), dim3(128), 0, stream>>>(big, input_ids, scale);

        // ctx[b,q,nh,d] = attn @ V   (batched, N=64) -> qw reused as ctx
        {
            dim3 g(1, 1, Bb * NHd);
            gemm_wmma_kernel<false, 0><<<g, blk256, 0, stream>>>(
                big, vw, nullptr, qw,
                Ss, DHd, Ss, Ss, Hd, Hd,
                (long)NHd * Ss * Ss, (long)Ss * Ss,       // A = attn
                (long)Ss * Hd, (long)DHd,                 // B = V
                (long)Ss * Hd, (long)DHd,                 // C = ctx (token-major, head slice)
                NHd);
        }

        // attention output projection + residual LN1
        launch_gemm_nn(qw, wo, bol, kw, Tn, Hd, Hd, Hd, Hd, Hd, 1, stream);
        ln_residual_kernel<<<dim3(Tn), blk256, 0, stream>>>(xw, kw, ln1_g + (long)l * Hd, ln1_b + (long)l * Hd, xw);

        // FFN: h = gelu(x@W1 + b1); x = LN(x + h@W2 + b2)
        launch_gemm_nn(xw, w1, b1l, big, Tn, FFd, Hd, Hd, FFd, FFd, 2, stream);
        launch_gemm_nn(big, w2, b2l, kw, Tn, Hd, FFd, FFd, Hd, Hd, 1, stream);
        ln_residual_kernel<<<dim3(Tn), blk256, 0, stream>>>(xw, kw, ln2_g + (long)l * Hd, ln2_b + (long)l * Hd, xw);
    }

    // ---- CLS rows -> [B,H] ----
    cls_extract_kernel<<<dim3((Bb * Hd) / 256), blk256, 0, stream>>>(xw, clsw);

    // ---- num = cls @ label_map^T  -> d_out[0 : B*NL] ----
    {
        dim3 g((NLd + 127) / 128, (Bb + 127) / 128, 1);
        gemm_wmma_kernel<true, 0><<<g, blk256, 0, stream>>>(
            clsw, label_map, nullptr, out,
            Bb, NLd, Hd, Hd, Hd, NLd,
            0, 0, 0, 0, 0, 0, 1);
    }

    // ---- norms, cosine scale, argmax ----
    row_norm_kernel<<<dim3(Bb),  blk256, 0, stream>>>(clsw,      cnw, Hd);
    row_norm_kernel<<<dim3(NLd), blk256, 0, stream>>>(label_map, lnw, Hd);
    cos_scale_kernel<<<dim3((Bb * NLd + 255) / 256), blk256, 0, stream>>>(out, cnw, lnw);
    argmax_kernel<<<dim3(Bb), blk256, 0, stream>>>(out, out + (long)Bb * NLd);
}
